// LinearAttention_14353780704014
// MI455X (gfx1250) — matrix operand order repaired and checked by reading, not compiled, not run on hardware
//
#include <hip/hip_runtime.h>
#include <math.h>

// ---------- problem constants ----------
#define Bc   8
#define Cc   256
#define Tc   8192
#define Hh   4
#define Dd   64
#define O3   768          // 3*CALC
#define EPSf 1e-5f

// ---------- types ----------
typedef __bf16 bf16;
typedef __attribute__((ext_vector_type(16))) __bf16 v16bf;
typedef __attribute__((ext_vector_type(8)))  __bf16 v8bf;
typedef __attribute__((ext_vector_type(8)))  float  v8f;
typedef __attribute__((ext_vector_type(4)))  int    v4i;

// ---------- async global->LDS copy (CDNA5 ASYNCcnt path) ----------
#if __has_builtin(__builtin_amdgcn_global_load_async_to_lds_b128)
#define HAS_ASYNC 1
#else
#define HAS_ASYNC 0
#endif

// copy 16 bytes from global to LDS (per-lane)
static __device__ inline void cp16(bf16* lds, const bf16* g) {
#if HAS_ASYNC
    __builtin_amdgcn_global_load_async_to_lds_b128(
        (__attribute__((address_space(1))) v4i*)(g),
        (__attribute__((address_space(3))) v4i*)(lds),
        /*imm offset=*/0, /*cpol=*/0);
#else
    *(v4i*)lds = *(const v4i*)g;
#endif
}

static __device__ inline void wait_async() {
#if HAS_ASYNC
#if __has_builtin(__builtin_amdgcn_s_wait_asynccnt)
    __builtin_amdgcn_s_wait_asynccnt(0);
#else
    asm volatile("s_wait_asynccnt 0x0" ::: "memory");
#endif
#endif
}

// ---------- WMMA helpers (wave32, 16x16x32 bf16 -> f32) ----------
static __device__ inline v8f wmma_bf16(v16bf a, v16bf b, v8f c) {
    return __builtin_amdgcn_wmma_f32_16x16x32_bf16(
        /*neg_a=*/false, a, /*neg_b=*/false, b,
        /*c_mod=*/(short)0, c, /*reuse_a=*/false, /*reuse_b=*/false);
}

// A fragment: 16x32 (MxK), row-major A with leading dim lda (elements).
// lane L (0..31): row = m0 + L%16 ; holds K = k0 + (L/16)*16 + {0..15}
static __device__ inline v16bf load_a_frag(const bf16* A, int lda, int m0, int k0, int lane) {
    const bf16* p = A + (m0 + (lane & 15)) * lda + k0 + ((lane >> 4) << 4);
    v8bf lo = *(const v8bf*)(p);
    v8bf hi = *(const v8bf*)(p + 8);
    v16bf a;
#pragma unroll
    for (int i = 0; i < 8; ++i) { a[i] = lo[i]; a[i + 8] = hi[i]; }
    return a;
}

// B fragment: 32x16 (KxN), sourced from Bt stored as [N][K] (K contiguous, ld=ldb).
// lane L: col = n0 + L%16 ; holds K = k0 + (L/16)*16 + {0..15}  -> one 32B load
static __device__ inline v16bf load_b_frag(const bf16* Bt, int ldb, int n0, int k0, int lane) {
    return *(const v16bf*)(Bt + (n0 + (lane & 15)) * ldb + k0 + ((lane >> 4) << 4));
}

// D fragment element (r=0..7): m = m0 + (L/16)*8 + r ; n = n0 + L%16

// ---------- kernel 0: convert weights f32 -> bf16 ----------
__global__ void k_cvt_weights(const float* Wpre, const float* Wproj,
                              bf16* Wpre_bf, bf16* Wproj_bf) {
    int i = blockIdx.x * blockDim.x + threadIdx.x;
    if (i < O3 * Cc)      Wpre_bf[i]  = (bf16)Wpre[i];
    if (i < Cc * Cc)      Wproj_bf[i] = (bf16)Wproj[i];
}

// ---------- kernel 1: transpose x[b] (C,T) f32 -> xt (T,C) bf16 ----------
__global__ void k_transpose_x(const float* __restrict__ x, bf16* __restrict__ xt) {
    __shared__ float tile[32][33];
    int tx = threadIdx.x, ty = threadIdx.y;
    int t = blockIdx.x * 32 + tx;   // T index (coalesced read)
    int c = blockIdx.y * 32 + ty;   // C index
    tile[ty][tx] = x[c * Tc + t];
    __syncthreads();
    int to = blockIdx.x * 32 + ty;
    int co = blockIdx.y * 32 + tx;
    xt[to * Cc + co] = (bf16)tile[tx][ty];
}

// ---------- kernel 2: prenet GEMM: h(768xT) = Wpre(768x256) @ x(256xT) ----------
// block: 256 thr (8 waves); block tile M=128 x N=128 ; wave: 16 x 128
// B-slab (128 cols x 256 K, 64 KB) staged once in LDS via async copies: 8x reuse.
__global__ void k_gemm_prenet(const bf16* __restrict__ W, const bf16* __restrict__ xt,
                              bf16* __restrict__ hbf, bf16* __restrict__ qT) {
    __shared__ __align__(32) bf16 sB[128 * 256];   // [col][k], 64 KB
    const int lane = threadIdx.x & 31, wave = threadIdx.x >> 5;
    const int m0 = blockIdx.y * 128 + wave * 16;
    const int n0 = blockIdx.x * 128;

    // stage B: 4096 x 16B chunks, 16 per thread
#pragma unroll
    for (int i = 0; i < 16; ++i) {
        int j   = threadIdx.x + i * 256;
        int col = j >> 5;            // 32 chunks per 256-elem row
        int cj  = j & 31;
        cp16(sB + j * 8, xt + (n0 + col) * Cc + cj * 8);
    }
    wait_async();
    __syncthreads();

    v8f acc[8];
#pragma unroll
    for (int i = 0; i < 8; ++i) acc[i] = (v8f)0.f;
#pragma unroll
    for (int k0 = 0; k0 < Cc; k0 += 32) {
        v16bf a = load_a_frag(W, Cc, m0, k0, lane);
#pragma unroll
        for (int ni = 0; ni < 8; ++ni)
            acc[ni] = wmma_bf16(a, load_b_frag(sB, Cc, ni * 16, k0, lane), acc[ni]);
    }
#pragma unroll
    for (int ni = 0; ni < 8; ++ni) {
#pragma unroll
        for (int r = 0; r < 8; ++r) {
            int m = m0 + ((lane >> 4) << 3) + r;
            int n = n0 + ni * 16 + (lane & 15);
            float v = acc[ni][r];
            hbf[m * Tc + n] = (bf16)v;
            int head = m / (3 * Dd), rm = m % (3 * Dd);
            if (rm < Dd)  // q slice -> transposed copy qT[head][t][d]
                qT[(head * Tc + n) * Dd + rm] = (bf16)v;
        }
    }
}

// ---------- kernel 3: softmax over T on k rows (in place, bf16) ----------
__global__ void k_softmax(bf16* __restrict__ hbf) {
    __shared__ float red[256];
    const int tid = threadIdx.x;
    const int head = blockIdx.x >> 6, d = blockIdx.x & 63;
    bf16* row = hbf + (head * (3 * Dd) + Dd + d) * Tc;

    float m = -3.0e38f;
    for (int i = tid; i < Tc; i += 256) m = fmaxf(m, (float)row[i]);
    red[tid] = m; __syncthreads();
    for (int s = 128; s > 0; s >>= 1) { if (tid < s) red[tid] = fmaxf(red[tid], red[tid + s]); __syncthreads(); }
    float M = red[0]; __syncthreads();

    float sum = 0.f;
    for (int i = tid; i < Tc; i += 256) sum += expf((float)row[i] - M);
    red[tid] = sum; __syncthreads();
    for (int s = 128; s > 0; s >>= 1) { if (tid < s) red[tid] += red[tid + s]; __syncthreads(); }
    float inv = 1.f / red[0];

    for (int i = tid; i < Tc; i += 256)
        row[i] = (bf16)(expf((float)row[i] - M) * inv);
}

// ---------- kernel 4: kv GEMM per head: kv[d][e] = sum_t k[d,t] v[e,t] ----------
// 4 blocks x 128 thr. k/v slabs (64 x 128 t each, 16 KB each) async-staged in LDS
// per 128-t step; all 4 waves share the v slab, each wave reads its k strip.
__global__ void k_gemm_kv(const bf16* __restrict__ hbf, bf16* __restrict__ kvT) {
    __shared__ __align__(32) bf16 sK[64 * 128];
    __shared__ __align__(32) bf16 sV[64 * 128];
    const int lane = threadIdx.x & 31, wave = threadIdx.x >> 5;
    const int head = blockIdx.x;
    const bf16* kbase = hbf + (head * (3 * Dd) + Dd) * Tc;       // k rows (softmaxed)
    const bf16* vbase = hbf + (head * (3 * Dd) + 2 * Dd) * Tc;   // v rows
    const int m0 = wave * 16;                                     // d strip
    v8f acc[4];
#pragma unroll
    for (int i = 0; i < 4; ++i) acc[i] = (v8f)0.f;

#pragma unroll 1
    for (int t0 = 0; t0 < Tc; t0 += 128) {
        __syncthreads();   // previous slab fully consumed
#pragma unroll
        for (int i = 0; i < 8; ++i) {
            int j  = threadIdx.x + i * 128;   // 1024 chunks per slab
            int e  = j >> 4;                  // 16 chunks per 128-elem row
            int cj = j & 15;
            cp16(sK + j * 8, kbase + e * Tc + t0 + cj * 8);
            cp16(sV + j * 8, vbase + e * Tc + t0 + cj * 8);
        }
        wait_async();
        __syncthreads();
#pragma unroll
        for (int kk = 0; kk < 128; kk += 32) {
            v16bf a = load_a_frag(sK, 128, m0, kk, lane);
#pragma unroll
            for (int ni = 0; ni < 4; ++ni)
                acc[ni] = wmma_bf16(a, load_b_frag(sV, 128, ni * 16, kk, lane), acc[ni]);
        }
    }
#pragma unroll
    for (int ni = 0; ni < 4; ++ni) {
#pragma unroll
        for (int r = 0; r < 8; ++r) {
            int d = m0 + ((lane >> 4) << 3) + r;
            int e = ni * 16 + (lane & 15);
            kvT[head * Dd * Dd + e * Dd + d] = (bf16)acc[ni][r];
        }
    }
}

// ---------- kernel 5: ctx GEMM: ctx[e,t] = sum_d kvT[e,d] * qT[t,d] ----------
// block 256 thr; block tile M=64 x N=128 ; writes ctxT[t][head*64+e] bf16
__global__ void k_gemm_ctx(const bf16* __restrict__ kvT, const bf16* __restrict__ qT,
                           bf16* __restrict__ ctxT) {
    const int lane = threadIdx.x & 31, wave = threadIdx.x >> 5;
    const int head = blockIdx.y;
    const bf16* A  = kvT + head * Dd * Dd;
    const bf16* Bt = qT + head * Tc * Dd;
    const int m0 = (wave & 3) * 16;
    const int n0 = blockIdx.x * 128 + (wave >> 2) * 64;
    v8f acc[4];
#pragma unroll
    for (int i = 0; i < 4; ++i) acc[i] = (v8f)0.f;
#pragma unroll
    for (int k0 = 0; k0 < Dd; k0 += 32) {
        v16bf a = load_a_frag(A, Dd, m0, k0, lane);
#pragma unroll
        for (int ni = 0; ni < 4; ++ni)
            acc[ni] = wmma_bf16(a, load_b_frag(Bt, Dd, n0 + ni * 16, k0, lane), acc[ni]);
    }
#pragma unroll
    for (int ni = 0; ni < 4; ++ni) {
#pragma unroll
        for (int r = 0; r < 8; ++r) {
            int e = m0 + ((lane >> 4) << 3) + r;
            int t = n0 + ni * 16 + (lane & 15);
            ctxT[t * Cc + head * Dd + e] = (bf16)acc[ni][r];
        }
    }
}

// ---------- kernel 6: proj GEMM + bias + scale + residual + LayerNorm ----------
// block 256 thr (8 waves) computes full 256-channel column block of 64 t's.
// B-slab (64 cols x 256 K, 32 KB) async-staged in LDS; LN tile in 64 KB LDS.
__global__ void k_gemm_proj_ln(const bf16* __restrict__ Wp, const bf16* __restrict__ ctxT,
                               const float* __restrict__ xres, const float* __restrict__ bproj,
                               const float* __restrict__ scale, const float* __restrict__ gamma,
                               const float* __restrict__ beta, float* __restrict__ out) {
    __shared__ float s[Cc][64];                    // 64 KB
    __shared__ __align__(32) bf16 sB[64 * 256];    // [col][k], 32 KB
    const int lane = threadIdx.x & 31, wave = threadIdx.x >> 5;
    const int t0 = blockIdx.x * 64;
    const int m0 = wave * 32;
    const float sc = scale[0];

    // stage B: 2048 x 16B chunks, 8 per thread
#pragma unroll
    for (int i = 0; i < 8; ++i) {
        int j   = threadIdx.x + i * 256;
        int col = j >> 5;
        int cj  = j & 31;
        cp16(sB + j * 8, ctxT + (t0 + col) * Cc + cj * 8);
    }
    wait_async();
    __syncthreads();

    v8f acc[2][4];
#pragma unroll
    for (int i = 0; i < 2; ++i)
#pragma unroll
        for (int j = 0; j < 4; ++j) acc[i][j] = (v8f)0.f;
#pragma unroll
    for (int k0 = 0; k0 < Cc; k0 += 32) {
        v16bf a0 = load_a_frag(Wp, Cc, m0, k0, lane);
        v16bf a1 = load_a_frag(Wp, Cc, m0 + 16, k0, lane);
#pragma unroll
        for (int ni = 0; ni < 4; ++ni) {
            v16bf b = load_b_frag(sB, Cc, ni * 16, k0, lane);
            acc[0][ni] = wmma_bf16(a0, b, acc[0][ni]);
            acc[1][ni] = wmma_bf16(a1, b, acc[1][ni]);
        }
    }
#pragma unroll
    for (int mi = 0; mi < 2; ++mi) {
#pragma unroll
        for (int ni = 0; ni < 4; ++ni) {
#pragma unroll
            for (int r = 0; r < 8; ++r) {
                int o  = m0 + mi * 16 + ((lane >> 4) << 3) + r;
                int tl = ni * 16 + (lane & 15);
                float v = sc * (acc[mi][ni][r] + bproj[o]) + xres[o * Tc + t0 + tl];
                s[o][tl] = v;
            }
        }
    }
    __syncthreads();
    if (threadIdx.x < 64) {
        int tl = threadIdx.x;
        float sum = 0.f, sq = 0.f;
#pragma unroll 4
        for (int c = 0; c < Cc; ++c) { float v = s[c][tl]; sum += v; sq += v * v; }
        float mean = sum * (1.f / Cc);
        float var  = sq * (1.f / Cc) - mean * mean;
        float rs   = rsqrtf(var + EPSf);
#pragma unroll 4
        for (int c = 0; c < Cc; ++c)
            out[c * Tc + t0 + tl] = (s[c][tl] - mean) * rs * gamma[c] + beta[c];
    }
}

// ---------- host launcher ----------
extern "C" void kernel_launch(void* const* d_in, const int* in_sizes, int n_in,
                              void* d_out, int out_size, void* d_ws, size_t ws_size,
                              hipStream_t stream) {
    const float* x     = (const float*)d_in[0];
    const float* Wpre  = (const float*)d_in[1];
    const float* Wproj = (const float*)d_in[2];
    const float* bproj = (const float*)d_in[3];
    const float* scale = (const float*)d_in[4];
    const float* gamma = (const float*)d_in[5];
    const float* beta  = (const float*)d_in[6];
    float* out = (float*)d_out;

    char* w = (char*)d_ws;
    bf16* Wpre_bf  = (bf16*)(w);                       // 768*256*2  = 393216
    bf16* Wproj_bf = (bf16*)(w + 393216);              // 256*256*2  = 131072
    bf16* xt       = (bf16*)(w + 524288);              // 8192*256*2 = 4 MiB
    bf16* hbf      = (bf16*)(w + 4718592);             // 768*8192*2 = 12 MiB
    bf16* qT       = (bf16*)(w + 17301504);            // 4*8192*64*2 = 4 MiB
    bf16* kvT      = (bf16*)(w + 21495808);            // 4*64*64*2  = 32 KB
    bf16* ctxT     = (bf16*)(w + 21528576);            // 8192*256*2 = 4 MiB

    k_cvt_weights<<<(O3 * Cc + 255) / 256, 256, 0, stream>>>(Wpre, Wproj, Wpre_bf, Wproj_bf);

    for (int b = 0; b < Bc; ++b) {
        const float* xb = x + (size_t)b * Cc * Tc;
        float* ob = out + (size_t)b * Cc * Tc;

        k_transpose_x<<<dim3(Tc / 32, Cc / 32), dim3(32, 32), 0, stream>>>(xb, xt);
        k_gemm_prenet<<<dim3(Tc / 128, O3 / 128), 256, 0, stream>>>(Wpre_bf, xt, hbf, qT);
        k_softmax<<<Hh * Dd, 256, 0, stream>>>(hbf);
        k_gemm_kv<<<Hh, 128, 0, stream>>>(hbf, kvT);
        k_gemm_ctx<<<dim3(Tc / 128, Hh), 256, 0, stream>>>(kvT, qT, ctxT);
        k_gemm_proj_ln<<<Tc / 64, 256, 0, stream>>>(Wproj_bf, ctxT, xb, bproj, scale, gamma, beta, ob);
    }
}
